// AttentionLayer_41601053229692
// MI455X (gfx1250) — compile-verified, matrix operand order checked by reading, not run on hardware
//
#include <hip/hip_runtime.h>
#include <hip/hip_bf16.h>

typedef __bf16 bf16_t;
typedef __attribute__((ext_vector_type(16))) __bf16 v16bf;
typedef __attribute__((ext_vector_type(8)))  float  v8f;

#define EMB 1024
#define SEQ 2048
#define HD  64
#define NH  16
#define BS  2
#define MTOT (BS * SEQ)      // 4096 rows
#define NQKV (3 * EMB)       // 3072

union FragBF16 {
    v16bf v;
    uint4 q[2];
};

static __device__ __forceinline__ v8f wmma_bf16(const v16bf& a, const v16bf& b, v8f c) {
    return __builtin_amdgcn_wmma_f32_16x16x32_bf16(false, a, false, b, (short)0, c, false, false);
}

// ---------------------------------------------------------------------------
// Stage 0: fp32 -> bf16 conversion (vectorized x4)
// ---------------------------------------------------------------------------
__global__ void cvt_f32_to_bf16(const float* __restrict__ src, bf16_t* __restrict__ dst, int n4) {
    int i = blockIdx.x * blockDim.x + threadIdx.x;
    if (i >= n4) return;
    float4 f = ((const float4*)src)[i];
    union { bf16_t e[4]; uint2 u; } p;
    p.e[0] = (bf16_t)f.x;
    p.e[1] = (bf16_t)f.y;
    p.e[2] = (bf16_t)f.z;
    p.e[3] = (bf16_t)f.w;
    ((uint2*)dst)[i] = p.u;
}

// ---------------------------------------------------------------------------
// Stage 1: QKV projection. C[m,f] = X[m,:] . W[f,:]  (torch linear layout)
// Block: 256 thr = 8 waves arranged 2(M) x 4(N); wave tile 64x32
// (4x2 WMMA tiles -> 0.75 fragment loads per WMMA, 4x B-fragment reuse).
// Epilogue scatters into Q[b,h,s,d], K[b,h,s,d], Vt[b,h,d,s].
// ---------------------------------------------------------------------------
__global__ __launch_bounds__(256) void qkv_gemm(const bf16_t* __restrict__ X,
                                                const bf16_t* __restrict__ W,
                                                bf16_t* __restrict__ Qo,
                                                bf16_t* __restrict__ Ko,
                                                bf16_t* __restrict__ Vt) {
    const int lane  = threadIdx.x & 31;
    const int wid   = threadIdx.x >> 5;
    const int waveM = wid >> 2;          // 0..1
    const int waveN = wid & 3;           // 0..3
    const int m0    = blockIdx.y * 128 + waveM * 64;
    const int n0    = blockIdx.x * 128 + waveN * 32;
    const int lh    = lane & 15;
    const int hi    = lane >> 4;
    const int kb    = hi * 8;            // per-lane K sub-chunk base (frag layout)

    const bf16_t* ap[4];
    const bf16_t* bp[2];
    #pragma unroll
    for (int mi = 0; mi < 4; ++mi) ap[mi] = X + (size_t)(m0 + mi * 16 + lh) * EMB;
    #pragma unroll
    for (int ni = 0; ni < 2; ++ni) bp[ni] = W + (size_t)(n0 + ni * 16 + lh) * EMB;

    v8f acc[4][2] = {};
    for (int k0 = 0; k0 < EMB; k0 += 32) {
        FragBF16 fb[2];
        #pragma unroll
        for (int ni = 0; ni < 2; ++ni) {
            fb[ni].q[0] = *(const uint4*)(bp[ni] + k0 + kb);
            fb[ni].q[1] = *(const uint4*)(bp[ni] + k0 + kb + 16);
        }
        #pragma unroll
        for (int mi = 0; mi < 4; ++mi) {
            FragBF16 fa;
            fa.q[0] = *(const uint4*)(ap[mi] + k0 + kb);
            fa.q[1] = *(const uint4*)(ap[mi] + k0 + kb + 16);
            acc[mi][0] = wmma_bf16(fa.v, fb[0].v, acc[mi][0]);
            acc[mi][1] = wmma_bf16(fa.v, fb[1].v, acc[mi][1]);
        }
    }

    #pragma unroll
    for (int mi = 0; mi < 4; ++mi) {
        #pragma unroll
        for (int ni = 0; ni < 2; ++ni) {
            int col = n0 + ni * 16 + lh;     // 0..3071
            int h   = col / 192;
            int r   = col % 192;
            #pragma unroll
            for (int rg = 0; rg < 8; ++rg) {
                int row = m0 + mi * 16 + rg + 8 * hi;   // 0..4095
                int b   = row >> 11;
                int s   = row & (SEQ - 1);
                bf16_t val = (bf16_t)acc[mi][ni][rg];
                size_t bh = (size_t)(b * NH + h);
                if (r < 64)
                    Qo[(bh * SEQ + s) * HD + r] = val;
                else if (r < 128)
                    Ko[(bh * SEQ + s) * HD + (r - 64)] = val;
                else
                    Vt[(bh * HD + (r - 128)) * SEQ + s] = val;
            }
        }
    }
}

// ---------------------------------------------------------------------------
// Stage 2: causal flash attention. One wave owns 16 query rows.
// S = Q.K^T via WMMA; online softmax (shuffle reductions within 16-lane
// halves match the C-matrix row layout); P staged through LDS to re-layout
// into A-fragments; O += P.V via WMMA with Vt rows as contiguous B-fragments.
// ---------------------------------------------------------------------------
__global__ __launch_bounds__(256) void attn_kernel(const bf16_t* __restrict__ Qg,
                                                   const bf16_t* __restrict__ Kg,
                                                   const bf16_t* __restrict__ Vtg,
                                                   bf16_t* __restrict__ Ctx) {
    __shared__ __align__(16) bf16_t pshare[8][16 * 64];

    const int lane = threadIdx.x & 31;
    const int wid  = threadIdx.x >> 5;
    const int task = blockIdx.x * 8 + wid;   // 0..4095
    const int mt   = task & 127;             // query tile index within (b,h)
    const int bh   = task >> 7;              // 0..31
    const int m0   = mt * 16;
    const int lh   = lane & 15;
    const int hi   = lane >> 4;
    const int kb   = hi * 8;

    const bf16_t* q  = Qg  + (size_t)bh * SEQ * HD;
    const bf16_t* k  = Kg  + (size_t)bh * SEQ * HD;
    const bf16_t* vt = Vtg + (size_t)bh * HD * SEQ;
    bf16_t* pl = &pshare[wid][0];

    // Q fragments (head_dim 64 = 2 WMMA k-steps), resident for whole loop
    FragBF16 qf[2];
    {
        const bf16_t* qrow = q + (size_t)(m0 + lh) * HD;
        qf[0].q[0] = *(const uint4*)(qrow + kb);
        qf[0].q[1] = *(const uint4*)(qrow + kb + 16);
        qf[1].q[0] = *(const uint4*)(qrow + 32 + kb);
        qf[1].q[1] = *(const uint4*)(qrow + 32 + kb + 16);
    }

    const float scale = 0.125f;           // 1/sqrt(64)
    const float l2e   = 1.44269504088896f;

    float mrow[8], lrow[8];
    v8f o[4] = {};
    #pragma unroll
    for (int r = 0; r < 8; ++r) { mrow[r] = -1e30f; lrow[r] = 0.0f; }

    for (int j0 = 0; j0 <= m0; j0 += 64) {
        // ---- S = Q.K^T over this 64-key block (4 n-tiles) ----
        v8f s[4];
        #pragma unroll
        for (int t = 0; t < 4; ++t) {
            const bf16_t* krow = k + (size_t)(j0 + t * 16 + lh) * HD;
            FragBF16 kf0, kf1;
            kf0.q[0] = *(const uint4*)(krow + kb);
            kf0.q[1] = *(const uint4*)(krow + kb + 16);
            kf1.q[0] = *(const uint4*)(krow + 32 + kb);
            kf1.q[1] = *(const uint4*)(krow + 32 + kb + 16);
            v8f z = {};
            z = wmma_bf16(qf[0].v, kf0.v, z);
            z = wmma_bf16(qf[1].v, kf1.v, z);
            s[t] = z;
        }

        // ---- scale + causal mask (only the diagonal-straddling block) ----
        const bool needMask = (j0 + 64 > m0);
        #pragma unroll
        for (int t = 0; t < 4; ++t) {
            #pragma unroll
            for (int r = 0; r < 8; ++r) {
                float v = s[t][r] * scale;
                if (needMask) {
                    int nn = j0 + t * 16 + lh;
                    int mm = m0 + r + 8 * hi;
                    v = (nn > mm) ? -3.0e38f : v;
                }
                s[t][r] = v;
            }
        }

        // ---- online softmax: row max over the 16-lane half ----
        float mnew[8];
        #pragma unroll
        for (int r = 0; r < 8; ++r) {
            float v = fmaxf(fmaxf(s[0][r], s[1][r]), fmaxf(s[2][r], s[3][r]));
            v = fmaxf(v, __shfl_xor(v, 1, 32));
            v = fmaxf(v, __shfl_xor(v, 2, 32));
            v = fmaxf(v, __shfl_xor(v, 4, 32));
            v = fmaxf(v, __shfl_xor(v, 8, 32));
            mnew[r] = fmaxf(mrow[r], v);
        }
        #pragma unroll
        for (int r = 0; r < 8; ++r) {
            float corr = exp2f((mrow[r] - mnew[r]) * l2e);
            lrow[r] *= corr;
            #pragma unroll
            for (int t = 0; t < 4; ++t) o[t][r] *= corr;
            mrow[r] = mnew[r];
        }

        // ---- P = exp(S - m), accumulate row sums ----
        float psum[8] = {};
        #pragma unroll
        for (int t = 0; t < 4; ++t) {
            #pragma unroll
            for (int r = 0; r < 8; ++r) {
                float p = exp2f((s[t][r] - mrow[r]) * l2e);
                s[t][r] = p;
                psum[r] += p;
            }
        }
        #pragma unroll
        for (int r = 0; r < 8; ++r) {
            float v = psum[r];
            v += __shfl_xor(v, 1, 32);
            v += __shfl_xor(v, 2, 32);
            v += __shfl_xor(v, 4, 32);
            v += __shfl_xor(v, 8, 32);
            lrow[r] += v;
        }

        // ---- stage P through LDS: C-layout -> row-major 16x64 bf16 ----
        #pragma unroll
        for (int t = 0; t < 4; ++t) {
            #pragma unroll
            for (int r = 0; r < 8; ++r) {
                int row = r + 8 * hi;
                int col = t * 16 + lh;
                pl[row * 64 + col] = (bf16_t)s[t][r];
            }
        }

        // ---- reload P as A-fragments (2 k-steps over 64 keys) ----
        FragBF16 pf[2];
        pf[0].q[0] = *(const uint4*)(pl + lh * 64 + kb);
        pf[0].q[1] = *(const uint4*)(pl + lh * 64 + kb + 16);
        pf[1].q[0] = *(const uint4*)(pl + lh * 64 + 32 + kb);
        pf[1].q[1] = *(const uint4*)(pl + lh * 64 + 32 + kb + 16);

        // ---- O += P.V  (B-fragments are contiguous rows of Vt) ----
        #pragma unroll
        for (int t = 0; t < 4; ++t) {
            const bf16_t* vrow = vt + (size_t)(t * 16 + lh) * SEQ + j0;
            FragBF16 vf0, vf1;
            vf0.q[0] = *(const uint4*)(vrow + kb);
            vf0.q[1] = *(const uint4*)(vrow + kb + 16);
            vf1.q[0] = *(const uint4*)(vrow + 32 + kb);
            vf1.q[1] = *(const uint4*)(vrow + 32 + kb + 16);
            o[t] = wmma_bf16(pf[0].v, vf0.v, o[t]);
            o[t] = wmma_bf16(pf[1].v, vf1.v, o[t]);
        }
    }

    // ---- finalize: O / l, write context [b, s, h*64+d] as bf16 ----
    const int b = bh >> 4;
    const int h = bh & 15;
    #pragma unroll
    for (int t = 0; t < 4; ++t) {
        #pragma unroll
        for (int r = 0; r < 8; ++r) {
            int row = m0 + r + 8 * hi;
            int d   = t * 16 + lh;
            float val = o[t][r] / lrow[r];
            Ctx[((size_t)(b * SEQ + row)) * EMB + h * HD + d] = (bf16_t)val;
        }
    }
}

// ---------------------------------------------------------------------------
// Stage 3: output projection + bias, fp32 result. Wave tile 64x32 (4x2).
// ---------------------------------------------------------------------------
__global__ __launch_bounds__(256) void out_gemm(const bf16_t* __restrict__ A,
                                                const bf16_t* __restrict__ W,
                                                const float* __restrict__ bias,
                                                float* __restrict__ Out) {
    const int lane  = threadIdx.x & 31;
    const int wid   = threadIdx.x >> 5;
    const int waveM = wid >> 2;          // 0..1
    const int waveN = wid & 3;           // 0..3
    const int m0    = blockIdx.y * 128 + waveM * 64;
    const int n0    = blockIdx.x * 128 + waveN * 32;
    const int lh    = lane & 15;
    const int hi    = lane >> 4;
    const int kb    = hi * 8;

    const bf16_t* ap[4];
    const bf16_t* bp[2];
    #pragma unroll
    for (int mi = 0; mi < 4; ++mi) ap[mi] = A + (size_t)(m0 + mi * 16 + lh) * EMB;
    #pragma unroll
    for (int ni = 0; ni < 2; ++ni) bp[ni] = W + (size_t)(n0 + ni * 16 + lh) * EMB;

    v8f acc[4][2] = {};
    for (int k0 = 0; k0 < EMB; k0 += 32) {
        FragBF16 fb[2];
        #pragma unroll
        for (int ni = 0; ni < 2; ++ni) {
            fb[ni].q[0] = *(const uint4*)(bp[ni] + k0 + kb);
            fb[ni].q[1] = *(const uint4*)(bp[ni] + k0 + kb + 16);
        }
        #pragma unroll
        for (int mi = 0; mi < 4; ++mi) {
            FragBF16 fa;
            fa.q[0] = *(const uint4*)(ap[mi] + k0 + kb);
            fa.q[1] = *(const uint4*)(ap[mi] + k0 + kb + 16);
            acc[mi][0] = wmma_bf16(fa.v, fb[0].v, acc[mi][0]);
            acc[mi][1] = wmma_bf16(fa.v, fb[1].v, acc[mi][1]);
        }
    }

    #pragma unroll
    for (int mi = 0; mi < 4; ++mi) {
        #pragma unroll
        for (int ni = 0; ni < 2; ++ni) {
            int   col = n0 + ni * 16 + lh;
            float bv  = bias[col];
            #pragma unroll
            for (int rg = 0; rg < 8; ++rg) {
                int row = m0 + mi * 16 + rg + 8 * hi;
                Out[(size_t)row * EMB + col] = acc[mi][ni][rg] + bv;
            }
        }
    }
}

// ---------------------------------------------------------------------------
extern "C" void kernel_launch(void* const* d_in, const int* in_sizes, int n_in,
                              void* d_out, int out_size, void* d_ws, size_t ws_size,
                              hipStream_t stream) {
    const float* x    = (const float*)d_in[0];   // [2,2048,1024]
    const float* wqkv = (const float*)d_in[1];   // [3072,1024]
    const float* wout = (const float*)d_in[2];   // [1024,1024]
    const float* bout = (const float*)d_in[3];   // [1024]
    float* out = (float*)d_out;                  // [2,2048,1024]

    char* ws = (char*)d_ws;
    bf16_t* xb    = (bf16_t*)(ws);                          // 8 MiB
    bf16_t* wqkvb = (bf16_t*)(ws + (8u  << 20));            // 6 MiB
    bf16_t* woutb = (bf16_t*)(ws + (14u << 20));            // 2 MiB
    bf16_t* Qb    = (bf16_t*)(ws + (16u << 20));            // 8 MiB
    bf16_t* Kb    = (bf16_t*)(ws + (24u << 20));            // 8 MiB
    bf16_t* Vtb   = (bf16_t*)(ws + (32u << 20));            // 8 MiB
    bf16_t* Ctx   = (bf16_t*)(ws + (40u << 20));            // 8 MiB

    const int nx = MTOT * EMB;        // 4194304
    const int nq = NQKV * EMB;        // 3145728
    const int nw = EMB * EMB;         // 1048576
    cvt_f32_to_bf16<<<(nx / 4 + 255) / 256, 256, 0, stream>>>(x,    xb,    nx / 4);
    cvt_f32_to_bf16<<<(nq / 4 + 255) / 256, 256, 0, stream>>>(wqkv, wqkvb, nq / 4);
    cvt_f32_to_bf16<<<(nw / 4 + 255) / 256, 256, 0, stream>>>(wout, woutb, nw / 4);

    qkv_gemm<<<dim3(NQKV / 128, MTOT / 128), 256, 0, stream>>>(xb, wqkvb, Qb, Kb, Vtb);
    attn_kernel<<<(BS * NH * (SEQ / 16)) / 8, 256, 0, stream>>>(Qb, Kb, Vtb, Ctx);
    out_gemm<<<dim3(EMB / 128, MTOT / 128), 256, 0, stream>>>(Ctx, woutb, bout, out);

    (void)in_sizes; (void)n_in; (void)out_size; (void)ws_size;
}
